// BLiqNet_59244778881188
// MI455X (gfx1250) — matrix operand order repaired and checked by reading, not compiled, not run on hardware
//
#include <hip/hip_runtime.h>

// ---------------------------------------------------------------------------
// Fused LiquidNet forward for MI455X (gfx1250, wave32, WMMA).
//
// Roofline: mandatory HBM traffic = x (512MB f32 in) + out (256MB f32 out)
// = 768MB -> ~33us @23.3TB/s. The 95 GFLOP of GEMM in bf16 WMMA (f32 accum)
// is ~20us, so the fused kernel is HBM-bound; f32 WMMA (16x16x4) would be
// ~10x above the memory floor. Weights (512KB f32) are converted to bf16 in
// LDS once per block; per-row state lives in VGPRs; D-layout -> A-layout
// transposes between chained matmuls use a 16x128 bf16 LDS tile/rowgroup.
// This revision targets VALU: f32->bf16 via compiler fptrunc (so ISel can
// select v_cvt_pk_bf16_f32) and hardware v_tanh_f32 (builtin or inline asm).
// ---------------------------------------------------------------------------

typedef __attribute__((ext_vector_type(16))) __bf16 v16bf;
typedef __attribute__((ext_vector_type(2)))  __bf16 v2bf;
typedef __attribute__((ext_vector_type(8)))  float  v8f;
typedef unsigned short u16b;

#define N_ROWS          262144
#define D_IN            512
#define HDIM            128
#define D_OUT           256
#define ROWS_PER_BLOCK  64
#define THREADS         256

// LDS: Wx 131072 + Wh 32768 + Wu 32768 + Wout 65536 + sH 16384 + biases 2048
#define SMEM_BYTES (131072 + 32768 + 32768 + 65536 + 16384 + 2048)

// ---- f32 -> bf16 conversions: let the compiler use native cvt insts ----
static __device__ __forceinline__ u16b f2b(float f) {
  __bf16 h = (__bf16)f;            // fptrunc, RNE; native cvt on gfx1250
  u16b r;
  __builtin_memcpy(&r, &h, 2);
  return r;
}
static __device__ __forceinline__ unsigned f2b2(float a, float b) {
#if __has_builtin(__builtin_amdgcn_cvt_pk_bf16_f32)
  v2bf t = __builtin_amdgcn_cvt_pk_bf16_f32(a, b);
#else
  v2bf t;
  t[0] = (__bf16)a;                // pair of fptruncs + build_vector:
  t[1] = (__bf16)b;                // ISel can fold into v_cvt_pk_bf16_f32
#endif
  unsigned r;
  __builtin_memcpy(&r, &t, 4);
  return r;
}
static __device__ __forceinline__ float b2f(u16b h) {
  return __uint_as_float(((unsigned)h) << 16);
}
// ---- hardware tanh (V_TANH_F32 is a gfx1250 TRANS op) ----
static __device__ __forceinline__ float fast_tanh(float x) {
#if __has_builtin(__builtin_amdgcn_tanhf)
  return __builtin_amdgcn_tanhf(x);
#else
  float r;
  // TRANS-op hazard rule (ISA 7.4): 1 independent op before result use.
  asm volatile("v_tanh_f32 %0, %1\n\tv_nop" : "=v"(r) : "v"(x));
  return r;
#endif
}

static __device__ __forceinline__ v8f wmma_bf16(v16bf a, v16bf b, v8f c) {
  // (neg_a, A, neg_b, B, c_mod, C, reuse_a, reuse_b)
  return __builtin_amdgcn_wmma_f32_16x16x32_bf16(false, a, false, b, (short)0, c,
                                                 false, false);
}

// B operand (32x16, K x N) from row-major bf16 [N][kdim] weight in LDS.
// lane: n = lane%16; chunks at K = k0 + 8*(lane/16) and +16 (8 bf16 each).
static __device__ __forceinline__ v16bf load_b_lds(const u16b* W, int kdim,
                                                   int ntile, int k0,
                                                   int laneM, int laneHalf) {
  const u16b* p = W + (ntile * 16 + laneM) * kdim + k0 + laneHalf * 8;
  v16bf b;
  __builtin_memcpy(&b, p, 16);
  __builtin_memcpy((char*)&b + 16, p + 16, 16);
  return b;
}

// A operand (16x32) from row-major bf16 [16][HDIM] tile in LDS.
// lane: m = lane%16; chunks at K = k0 + 8*(lane/16) and +16 (8 bf16 each).
static __device__ __forceinline__ v16bf load_a_lds(const u16b* sHg, int k0,
                                                   int laneM, int laneHalf) {
  const u16b* p = sHg + laneM * HDIM + k0 + laneHalf * 8;
  v16bf a;
  __builtin_memcpy(&a, p, 16);
  __builtin_memcpy((char*)&a + 16, p + 16, 16);
  return a;
}

// A operand built directly from global f32 x (stream), packed-converted.
static __device__ __forceinline__ v16bf load_a_x(const float* xrow, int k0,
                                                 int laneHalf) {
  const float* p = xrow + k0 + laneHalf * 8;
  union { v16bf v; unsigned w[8]; } A;
#pragma unroll
  for (int i = 0; i < 4; ++i) {
    A.w[i]     = f2b2(p[2 * i],      p[2 * i + 1]);
    A.w[4 + i] = f2b2(p[16 + 2 * i], p[16 + 2 * i + 1]);
  }
  return A.v;
}

__global__ __launch_bounds__(THREADS)
void liquidnet_fused(const float* __restrict__ x,   const float* __restrict__ Wx,
                     const float* __restrict__ bx,  const float* __restrict__ Wh,
                     const float* __restrict__ Wu,  const float* __restrict__ b_ode,
                     const float* __restrict__ Wout,const float* __restrict__ bout,
                     float* __restrict__ out) {
  extern __shared__ unsigned char smem_raw[];
  u16b* sWx   = (u16b*)smem_raw;                 // 128*512
  u16b* sWh   = sWx   + HDIM * D_IN;             // 128*128
  u16b* sWu   = sWh   + HDIM * HDIM;             // 128*128
  u16b* sWout = sWu   + HDIM * HDIM;             // 256*128
  u16b* sH    = sWout + D_OUT * HDIM;            // 4 * 16*128
  float* sBx   = (float*)(sH + 4 * 16 * HDIM);   // 128
  float* sBode = sBx + HDIM;                     // 128
  float* sBout = sBode + HDIM;                   // 256

  const int tid = threadIdx.x;

  // ---- Stage weights f32 -> bf16 into LDS (cooperative, packed cvt) ----
  for (int i = tid * 4; i < HDIM * D_IN; i += THREADS * 4) {
    float4 v = *(const float4*)(Wx + i);
    *(unsigned*)(sWx + i)     = f2b2(v.x, v.y);
    *(unsigned*)(sWx + i + 2) = f2b2(v.z, v.w);
  }
  for (int i = tid * 4; i < HDIM * HDIM; i += THREADS * 4) {
    float4 v = *(const float4*)(Wh + i);
    *(unsigned*)(sWh + i)     = f2b2(v.x, v.y);
    *(unsigned*)(sWh + i + 2) = f2b2(v.z, v.w);
    float4 w = *(const float4*)(Wu + i);
    *(unsigned*)(sWu + i)     = f2b2(w.x, w.y);
    *(unsigned*)(sWu + i + 2) = f2b2(w.z, w.w);
  }
  for (int i = tid * 4; i < D_OUT * HDIM; i += THREADS * 4) {
    float4 v = *(const float4*)(Wout + i);
    *(unsigned*)(sWout + i)     = f2b2(v.x, v.y);
    *(unsigned*)(sWout + i + 2) = f2b2(v.z, v.w);
  }
  if (tid < HDIM) { sBx[tid] = bx[tid]; sBode[tid] = b_ode[tid]; }
  sBout[tid] = bout[tid];  // THREADS == D_OUT
  __syncthreads();

  // ---- 8 waves = 4 rowgroups x 2 column-halves ----
  const int wave = tid >> 5, lane = tid & 31;
  const int rg = wave >> 1, half = wave & 1;
  const int laneM = lane & 15, laneHalf = lane >> 4;
  const int rowBase = blockIdx.x * ROWS_PER_BLOCK + rg * 16;
  u16b* sHg = sH + rg * 16 * HDIM;
  const int colBaseH = half * 64;

  // ---- u = x @ Wx^T + bx  (K = 512: 16 WMMA K-steps x 4 N-tiles) ----
  v8f h0[4];
#pragma unroll
  for (int t = 0; t < 4; ++t) {
    float bv = sBx[colBaseH + t * 16 + laneM];
#pragma unroll
    for (int r = 0; r < 8; ++r) h0[t][r] = bv;
  }
  const float* xrow = x + (size_t)(rowBase + laneM) * D_IN;
#pragma unroll
  for (int kk = 0; kk < 16; ++kk) {
    v16bf a = load_a_x(xrow, kk * 32, laneHalf);
    if (kk + 2 < 16) __builtin_prefetch(xrow + (kk + 2) * 32, 0, 1);
#pragma unroll
    for (int t = 0; t < 4; ++t) {
      v16bf b = load_b_lds(sWx, D_IN, half * 4 + t, kk * 32, laneM, laneHalf);
      h0[t] = wmma_bf16(a, b, h0[t]);
    }
  }

  // Stage h0 (bf16) into sH: A-source for uu, and h_arg for RK4 stage 1.
#pragma unroll
  for (int t = 0; t < 4; ++t) {
    int col = colBaseH + t * 16 + laneM;
#pragma unroll
    for (int r = 0; r < 8; ++r)
      sHg[(r + 8 * laneHalf) * HDIM + col] = f2b(h0[t][r]);
  }
  __syncthreads();

  // ---- uu = u @ Wu^T + b_ode ----
  v8f uu[4];
#pragma unroll
  for (int t = 0; t < 4; ++t) {
    float bv = sBode[colBaseH + t * 16 + laneM];
#pragma unroll
    for (int r = 0; r < 8; ++r) uu[t][r] = bv;
  }
#pragma unroll
  for (int kk = 0; kk < 4; ++kk) {
    v16bf a = load_a_lds(sHg, kk * 32, laneM, laneHalf);
#pragma unroll
    for (int t = 0; t < 4; ++t) {
      v16bf b = load_b_lds(sWu, HDIM, half * 4 + t, kk * 32, laneM, laneHalf);
      uu[t] = wmma_bf16(a, b, uu[t]);
    }
  }

  // ---- RK4: f(h) = -h + tanh(h @ Wh^T + uu), dt = 2 ----
  v8f hout[4];
  const float cvals[3] = {1.0f, 1.0f, 2.0f};                    // 0.5dt,0.5dt,dt
  const float wvals[4] = {2.0f/6.0f, 4.0f/6.0f, 4.0f/6.0f, 2.0f/6.0f}; // dt/6 wts
#pragma unroll
  for (int s = 0; s < 4; ++s) {
    // g = uu + h_arg @ Wh^T ; first K-step consumes uu as the C operand so
    // no accumulator copy is needed (D is written to fresh registers).
    v8f g[4];
    {
      v16bf a0 = load_a_lds(sHg, 0, laneM, laneHalf);
#pragma unroll
      for (int t = 0; t < 4; ++t) {
        v16bf b = load_b_lds(sWh, HDIM, half * 4 + t, 0, laneM, laneHalf);
        g[t] = wmma_bf16(a0, b, uu[t]);
      }
    }
#pragma unroll
    for (int kk = 1; kk < 4; ++kk) {
      v16bf a = load_a_lds(sHg, kk * 32, laneM, laneHalf);
#pragma unroll
      for (int t = 0; t < 4; ++t) {
        v16bf b = load_b_lds(sWh, HDIM, half * 4 + t, kk * 32, laneM, laneHalf);
        g[t] = wmma_bf16(a, b, g[t]);
      }
    }
    // k_s = -h_arg + tanh(g); accumulate hout; build next h_arg.
    v8f hnext[4];
#pragma unroll
    for (int t = 0; t < 4; ++t) {
      int col = colBaseH + t * 16 + laneM;
#pragma unroll
      for (int r = 0; r < 8; ++r) {
        float harg = b2f(sHg[(r + 8 * laneHalf) * HDIM + col]);
        float kv = -harg + fast_tanh(g[t][r]);
        if (s == 0) hout[t][r] = h0[t][r] + wvals[0] * kv;  // fold init
        else        hout[t][r] += wvals[s] * kv;
        if (s < 3)  hnext[t][r] = h0[t][r] + cvals[s] * kv;
      }
    }
    if (s < 3) {
      __syncthreads();  // all waves done reading current h_arg
#pragma unroll
      for (int t = 0; t < 4; ++t) {
        int col = colBaseH + t * 16 + laneM;
#pragma unroll
        for (int r = 0; r < 8; ++r)
          sHg[(r + 8 * laneHalf) * HDIM + col] = f2b(hnext[t][r]);
      }
      __syncthreads();  // next h_arg visible to the wave pair
    }
  }

  // ---- Stage final h into sH for the head matmul ----
  __syncthreads();
#pragma unroll
  for (int t = 0; t < 4; ++t) {
    int col = colBaseH + t * 16 + laneM;
#pragma unroll
    for (int r = 0; r < 8; ++r)
      sHg[(r + 8 * laneHalf) * HDIM + col] = f2b(hout[t][r]);
  }
  __syncthreads();

  // ---- out = h @ Wout^T + bout  (this wave: 128 output cols = 8 N-tiles) ----
  v16bf aH[4];
#pragma unroll
  for (int kk = 0; kk < 4; ++kk) aH[kk] = load_a_lds(sHg, kk * 32, laneM, laneHalf);
  const int colBaseO = half * 128;
#pragma unroll
  for (int t8 = 0; t8 < 8; ++t8) {
    int col = colBaseO + t8 * 16 + laneM;
    float bv = sBout[col];
    v8f acc;
#pragma unroll
    for (int r = 0; r < 8; ++r) acc[r] = bv;
#pragma unroll
    for (int kk = 0; kk < 4; ++kk) {
      v16bf b = load_b_lds(sWout, HDIM, half * 8 + t8, kk * 32, laneM, laneHalf);
      acc = wmma_bf16(aH[kk], b, acc);
    }
#pragma unroll
    for (int r = 0; r < 8; ++r)
      out[(size_t)(rowBase + r + 8 * laneHalf) * D_OUT + col] = acc[r];
  }
}

extern "C" void kernel_launch(void* const* d_in, const int* in_sizes, int n_in,
                              void* d_out, int out_size, void* d_ws, size_t ws_size,
                              hipStream_t stream) {
  (void)in_sizes; (void)n_in; (void)out_size; (void)d_ws; (void)ws_size;
  const float* x     = (const float*)d_in[0];
  const float* Wx    = (const float*)d_in[1];
  const float* bx    = (const float*)d_in[2];
  const float* Wh    = (const float*)d_in[3];
  const float* Wu    = (const float*)d_in[4];
  const float* b_ode = (const float*)d_in[5];
  const float* Wout  = (const float*)d_in[6];
  const float* bout  = (const float*)d_in[7];
  float* out = (float*)d_out;

  // CDNA5 allows up to 320KB LDS per workgroup; raise the dynamic-LDS cap.
  (void)hipFuncSetAttribute((const void*)liquidnet_fused,
                            hipFuncAttributeMaxDynamicSharedMemorySize,
                            (int)SMEM_BYTES);

  dim3 grid(N_ROWS / ROWS_PER_BLOCK);   // 4096 blocks
  dim3 block(THREADS);                  // 8 wave32 waves
  liquidnet_fused<<<grid, block, SMEM_BYTES, stream>>>(x, Wx, bx, Wh, Wu, b_ode,
                                                       Wout, bout, out);
}
